// SplitHeadAttention_37752762531952
// MI455X (gfx1250) — matrix operand order repaired and checked by reading, not compiled, not run on hardware
//
#include <hip/hip_runtime.h>
#include <hip/hip_bf16.h>

typedef __attribute__((ext_vector_type(16))) _Float16 v16h;
typedef __attribute__((ext_vector_type(8)))  float    v8f;

#define TT    4096
#define BB    8
#define HD    64
#define DM    128
#define LOG2E 1.44269504f

// ---------------------------------------------------------------------------
// Fragment loaders per CDNA5 ISA 7.12.2 VGPR layouts (wave32).
// A (16x32 f16): lane 0-15 -> M=lane, VGPR v holds K pair; lanes 16-31 same M,
//   upper K half of each 8-K group.  K pair base = (v/4)*16 + (lane/16)*8 + (v%4)*2.
// B (32x16 f16): lane -> N=lane%16, VGPR v holds K pair = (v/4)*16 + (lane/16)*8 + (v%4)*2.
// C/D (16x16 f32): VGPR r, lane l -> M = r + 8*(l/16), N = l%16.
// ---------------------------------------------------------------------------
__device__ __forceinline__ v16h load_frag_A(const _Float16* __restrict__ src,
                                            int rs, int row0, int k0) {
  int lane = threadIdx.x & 31;
  int row  = row0 + (lane & 15);
  int kh8  = (lane >> 4) << 3;
  union { v16h v; unsigned u[8]; } f;
  const _Float16* p = src + (size_t)row * rs + k0;
#pragma unroll
  for (int v = 0; v < 8; ++v) {
    int kb = ((v >> 2) << 4) + kh8 + ((v & 3) << 1);
    f.u[v] = *(const unsigned*)(p + kb);
  }
  return f.v;
}

// src laid out N-major / K-contiguous: element(k,n) at src[n*rs + k]
__device__ __forceinline__ v16h load_frag_B(const _Float16* __restrict__ src,
                                            int rs, int n0, int k0) {
  int lane = threadIdx.x & 31;
  int n    = n0 + (lane & 15);
  int khb  = (lane >> 4) << 3;
  union { v16h v; unsigned u[8]; } f;
  const _Float16* p = src + (size_t)n * rs + k0 + khb;
#pragma unroll
  for (int v = 0; v < 8; ++v) f.u[v] = *(const unsigned*)(p + ((v >> 2) << 4) + ((v & 3) << 1));
  return f.v;
}

__device__ __forceinline__ v8f wmma_f16(v16h a, v16h b, v8f c) {
  return __builtin_amdgcn_wmma_f32_16x16x32_f16(false, a, false, b,
                                                (short)0, c, false, false);
}

// CDNA5 async global->LDS copy (ASYNCcnt-tracked).  vdst = LDS byte offset
// (low 32 bits of the generic shared pointer, per ISA 10.2 LDS aperture),
// vaddr = 64-bit global address, GV mode.
__device__ __forceinline__ void async_load_b128(unsigned lds_off, const void* g) {
  asm volatile("global_load_async_to_lds_b128 %0, %1, off"
               :: "v"(lds_off), "v"(g) : "memory");
}

// ---------------------------------------------------------------------------
// Kernel 1: rank-3 projections -> f16 Q (pre-scaled by 1/8), K row-major,
// V transposed [d][t].  One block per (b,t), 64 threads = head dim.
// ---------------------------------------------------------------------------
__global__ __launch_bounds__(64) void proj_kernel(
    const float* __restrict__ x,
    const float* __restrict__ Wq1, const float* __restrict__ Wk1,
    const float* __restrict__ Wv1, const float* __restrict__ Wq2,
    const float* __restrict__ Wk2, const float* __restrict__ Wv2,
    _Float16* __restrict__ Q, _Float16* __restrict__ K,
    _Float16* __restrict__ Vt) {
  int bt = blockIdx.x;
  int d  = threadIdx.x;
  int b  = bt >> 12;          // /T
  int t  = bt & (TT - 1);
  const float* xr = x + (size_t)bt * 6;
  float x0 = xr[0], x1 = xr[1], x2 = xr[2];
  float x3 = xr[3], x4 = xr[4], x5 = xr[5];

  float q1 = x0 * Wq1[d] + x1 * Wq1[HD + d] + x2 * Wq1[2 * HD + d];
  float k1 = x0 * Wk1[d] + x1 * Wk1[HD + d] + x2 * Wk1[2 * HD + d];
  float v1 = x0 * Wv1[d] + x1 * Wv1[HD + d] + x2 * Wv1[2 * HD + d];
  float q2 = x3 * Wq2[d] + x4 * Wq2[HD + d] + x5 * Wq2[2 * HD + d];
  float k2 = x3 * Wk2[d] + x4 * Wk2[HD + d] + x5 * Wk2[2 * HD + d];
  float v2 = x3 * Wv2[d] + x4 * Wv2[HD + d] + x5 * Wv2[2 * HD + d];

  size_t r0 = ((size_t)(b * 2 + 0) * TT + t) * HD + d;
  size_t r1 = ((size_t)(b * 2 + 1) * TT + t) * HD + d;
  Q[r0] = (_Float16)(q1 * 0.125f);  K[r0] = (_Float16)k1;
  Q[r1] = (_Float16)(q2 * 0.125f);  K[r1] = (_Float16)k2;
  Vt[((size_t)(b * 2 + 0) * HD + d) * TT + t] = (_Float16)v1;
  Vt[((size_t)(b * 2 + 1) * HD + d) * TT + t] = (_Float16)v2;
}

// ---------------------------------------------------------------------------
// Kernel 2: Wout (128x128 f32, [k][n]) -> WoutT (f16, [n][k])
// ---------------------------------------------------------------------------
__global__ __launch_bounds__(256) void woutT_kernel(const float* __restrict__ Wout,
                                                    _Float16* __restrict__ WoutT) {
  int idx = blockIdx.x * 256 + threadIdx.x;     // 16384 elements
  int n = idx >> 7, k = idx & 127;
  WoutT[idx] = (_Float16)Wout[k * DM + n];
}

// ---------------------------------------------------------------------------
// Kernel 3: causal flash attention. grid = (T/64, B*2 heads), 128 threads.
// Workgroup cooperatively streams 32-key K/V blocks into double-buffered LDS
// via global_load_async_to_lds_b128; each wave owns a 16-row Q tile and runs
// online softmax + WMMA out of LDS.
// ---------------------------------------------------------------------------
__global__ __launch_bounds__(128) void attn_kernel(
    const _Float16* __restrict__ Q, const _Float16* __restrict__ K,
    const _Float16* __restrict__ Vt, _Float16* __restrict__ AO) {
  __shared__ _Float16 ldsK[2][32 * HD];   // [key][d], rs=64   (4 KB each)
  __shared__ _Float16 ldsV[2][HD * 32];   // [d][key], rs=32   (4 KB each)
  __shared__ _Float16 ldsP[4][16 * 32];   // per-wave P tile

  int tid   = threadIdx.x;
  int lane  = tid & 31;
  int wave  = tid >> 5;
  int kh    = lane >> 4;
  int ln    = lane & 15;
  int bh    = blockIdx.y;
  int qbase = blockIdx.x * 64 + wave * 16;

  const _Float16* Qp = Q  + (size_t)bh * TT * HD;
  const _Float16* Kp = K  + (size_t)bh * TT * HD;
  const _Float16* Vp = Vt + (size_t)bh * HD * TT;

  // causal bounds: per-wave compute bound, block-wide streaming bound
  int myend = ((qbase + 47) >> 5) << 5;
  int qmax  = blockIdx.x * 64 + 48;
  int send  = ((qmax + 47) >> 5) << 5;

  // per-thread async-copy addressing (K: 4KB contiguous; V: 64 rows x 64B)
  int vrow = tid >> 1, vofs = (tid & 1) * 32;
  unsigned kdst0 = ((unsigned)(size_t)(void*)&ldsK[0][0]) + (unsigned)tid * 16u;
  unsigned kdst1 = ((unsigned)(size_t)(void*)&ldsK[1][0]) + (unsigned)tid * 16u;
  unsigned vdst0 = ((unsigned)(size_t)(void*)&ldsV[0][0]) + (unsigned)(vrow * 64 + vofs);
  unsigned vdst1 = ((unsigned)(size_t)(void*)&ldsV[1][0]) + (unsigned)(vrow * 64 + vofs);

  // prologue: stream block 0 into buffer 0
  {
    const char* kg = (const char*)(Kp) + tid * 16;
    const char* vg = (const char*)(Vp + (size_t)vrow * TT) + vofs;
    async_load_b128(kdst0, kg);
    async_load_b128(kdst0 + 2048u, kg + 2048);
    async_load_b128(vdst0, vg);
    async_load_b128(vdst0 + 16u, vg + 16);
  }

  v16h aq0 = load_frag_A(Qp, HD, qbase, 0);
  v16h aq1 = load_frag_A(Qp, HD, qbase, 32);

  v8f o0 = {}, o1 = {}, o2 = {}, o3 = {};
  float m_[8], l_[8];
#pragma unroll
  for (int r = 0; r < 8; ++r) { m_[r] = -1e30f; l_[r] = 0.0f; }

  int bi = 0;
  for (int s = 0; s < send; s += 32) {
    asm volatile("s_wait_asynccnt 0x0" ::: "memory");
    __syncthreads();   // all waves' tiles resident; all done with other buffer
    if (s + 32 < send) {
      unsigned kdst = bi ? kdst0 : kdst1;
      unsigned vdst = bi ? vdst0 : vdst1;
      const char* kg = (const char*)(Kp + (size_t)(s + 32) * HD) + tid * 16;
      const char* vg = (const char*)(Vp + (size_t)vrow * TT + (s + 32)) + vofs;
      async_load_b128(kdst, kg);
      async_load_b128(kdst + 2048u, kg + 2048);
      async_load_b128(vdst, vg);
      async_load_b128(vdst + 16u, vg + 16);
    }

    if (s < myend) {   // wave-uniform predicate: EXEC stays all-ones
      const _Float16* kb = ldsK[bi];
      const _Float16* vb = ldsV[bi];

      // ---- S = Q(16x64) . K^T(64x32): two 16-key tiles, K split 32+32
      v8f s0 = {}, s1 = {};
      {
        v16h bk;
        bk = load_frag_B(kb, HD, 0, 0);   s0 = wmma_f16(aq0, bk, s0);
        bk = load_frag_B(kb, HD, 0, 32);  s0 = wmma_f16(aq1, bk, s0);
        bk = load_frag_B(kb, HD, 16, 0);  s1 = wmma_f16(aq0, bk, s1);
        bk = load_frag_B(kb, HD, 16, 32); s1 = wmma_f16(aq1, bk, s1);
      }
      // ---- causal mask only on diagonal-intersecting blocks (uniform branch)
      if (s + 31 > qbase) {
        int col0 = s + ln, col1 = s + 16 + ln;
#pragma unroll
        for (int r = 0; r < 8; ++r) {
          int row = qbase + r + 8 * kh;
          if (col0 > row) s0[r] = -1e30f;
          if (col1 > row) s1[r] = -1e30f;
        }
      }
      // ---- online softmax across the 16 lanes of each half-wave
      float alpha[8];
#pragma unroll
      for (int r = 0; r < 8; ++r) {
        float mx = fmaxf(s0[r], s1[r]);
#pragma unroll
        for (int msk = 1; msk < 16; msk <<= 1)
          mx = fmaxf(mx, __shfl_xor(mx, msk, 32));
        float mn = fmaxf(m_[r], mx);
        alpha[r] = exp2f((m_[r] - mn) * LOG2E);
        m_[r]    = mn;
      }
#pragma unroll
      for (int r = 0; r < 8; ++r) {
        float p0 = exp2f((s0[r] - m_[r]) * LOG2E);
        float p1 = exp2f((s1[r] - m_[r]) * LOG2E);
        s0[r] = p0; s1[r] = p1;
        float rsum = p0 + p1;
#pragma unroll
        for (int msk = 1; msk < 16; msk <<= 1)
          rsum += __shfl_xor(rsum, msk, 32);
        l_[r] = l_[r] * alpha[r] + rsum;
        o0[r] *= alpha[r]; o1[r] *= alpha[r];
        o2[r] *= alpha[r]; o3[r] *= alpha[r];
      }
      // ---- P: C-layout -> A-layout via per-wave LDS tile
      _Float16* lp = ldsP[wave];
#pragma unroll
      for (int r = 0; r < 8; ++r) {
        int row = r + 8 * kh;
        lp[row * 32 + ln]      = (_Float16)s0[r];
        lp[row * 32 + 16 + ln] = (_Float16)s1[r];
      }
      asm volatile("s_wait_dscnt 0x0" ::: "memory");
      v16h pf = load_frag_A(lp, 32, 0, 0);
      // ---- O(16x64) += P(16x32) . V(32x64)   (V tile from LDS, rs=32)
      o0 = wmma_f16(pf, load_frag_B(vb, 32, 0,  0), o0);
      o1 = wmma_f16(pf, load_frag_B(vb, 32, 16, 0), o1);
      o2 = wmma_f16(pf, load_frag_B(vb, 32, 32, 0), o2);
      o3 = wmma_f16(pf, load_frag_B(vb, 32, 48, 0), o3);
    }
    bi ^= 1;
  }

  // ---- normalize + write concat(attn1, attn2) tile as f16
  int b = bh >> 1, h = bh & 1;
#pragma unroll
  for (int r = 0; r < 8; ++r) {
    float inv = 1.0f / l_[r];
    int trow  = qbase + r + 8 * kh;
    _Float16* op = AO + ((size_t)b * TT + trow) * DM + h * HD;
    op[ln]      = (_Float16)(o0[r] * inv);
    op[16 + ln] = (_Float16)(o1[r] * inv);
    op[32 + ln] = (_Float16)(o2[r] * inv);
    op[48 + ln] = (_Float16)(o3[r] * inv);
  }
}

// ---------------------------------------------------------------------------
// Kernel 4: out = AO (32768x128 f16) . Wout (128x128) -> f32.
// One 16x16 output tile per wave; K=128 as 4 k=32 WMMA steps.
// ---------------------------------------------------------------------------
__global__ __launch_bounds__(256) void out_gemm_kernel(
    const _Float16* __restrict__ AO, const _Float16* __restrict__ WoutT,
    float* __restrict__ out) {
  int wave = threadIdx.x >> 5;
  int lane = threadIdx.x & 31;
  int id   = blockIdx.x * 8 + wave;
  int mt   = id >> 3;
  int nt   = id & 7;

  v8f c = {};
#pragma unroll
  for (int kk = 0; kk < 4; ++kk) {
    v16h a = load_frag_A(AO,    DM, mt * 16, kk * 32);
    v16h b = load_frag_B(WoutT, DM, nt * 16, kk * 32);
    c = wmma_f16(a, b, c);
  }
  int col = nt * 16 + (lane & 15);
#pragma unroll
  for (int r = 0; r < 8; ++r) {
    int row = mt * 16 + r + 8 * (lane >> 4);
    out[(size_t)row * DM + col] = c[r];
  }
}

// ---------------------------------------------------------------------------
extern "C" void kernel_launch(void* const* d_in, const int* in_sizes, int n_in,
                              void* d_out, int out_size, void* d_ws, size_t ws_size,
                              hipStream_t stream) {
  const float* x    = (const float*)d_in[0];
  const float* Wq1  = (const float*)d_in[1];
  const float* Wk1  = (const float*)d_in[2];
  const float* Wv1  = (const float*)d_in[3];
  const float* Wq2  = (const float*)d_in[4];
  const float* Wk2  = (const float*)d_in[5];
  const float* Wv2  = (const float*)d_in[6];
  const float* Wout = (const float*)d_in[7];

  const size_t SEG = (size_t)BB * 2 * TT * HD * sizeof(_Float16);  // 8 MB
  char* ws = (char*)d_ws;
  _Float16* Q     = (_Float16*)(ws);
  _Float16* K     = (_Float16*)(ws + SEG);
  _Float16* Vt    = (_Float16*)(ws + 2 * SEG);
  _Float16* AO    = (_Float16*)(ws + 3 * SEG);
  _Float16* WoutT = (_Float16*)(ws + 4 * SEG);
  float* out = (float*)d_out;

  proj_kernel<<<BB * TT, 64, 0, stream>>>(x, Wq1, Wk1, Wv1, Wq2, Wk2, Wv2,
                                          Q, K, Vt);
  woutT_kernel<<<64, 256, 0, stream>>>(Wout, WoutT);
  attn_kernel<<<dim3(TT / 64, BB * 2), 128, 0, stream>>>(Q, K, Vt, AO);
  out_gemm_kernel<<<(BB * TT / 16), 256, 0, stream>>>(AO, WoutT, out);
}
